// HardNegativeInfoNCELoss_68693706932261
// MI455X (gfx1250) — matrix operand order repaired and checked by reading, not compiled, not run on hardware
//
#include <hip/hip_runtime.h>
#include <hip/hip_bf16.h>
#include <math.h>
#include <stdint.h>

#define N_ROWS 8192
#define DIM    256
#define TILE_H (16 * DIM)            // one B tile: 16 z-rows of 256 halfs
#define INV_T  (1.0f / 0.07f)

typedef __attribute__((ext_vector_type(16))) _Float16 v16h;
typedef __attribute__((ext_vector_type(8)))  _Float16 v8h;
typedef __attribute__((ext_vector_type(8)))  float    v8f;
typedef __attribute__((ext_vector_type(4)))  int      v4i;

typedef __attribute__((address_space(1))) v4i* gptr_b128;   // global b128
typedef __attribute__((address_space(3))) v4i* lptr_b128;   // LDS b128

#if __has_builtin(__builtin_amdgcn_global_load_async_to_lds_b128)
#define ASYNC_STAGE 1
#else
#define ASYNC_STAGE 0
#endif

// ---------------------------------------------------------------------------
// Deterministic per-(i,j) Gumbel via integer hash (stand-in for jax threefry)
// ---------------------------------------------------------------------------
__device__ __forceinline__ float gumbel_hash(uint32_t i, uint32_t j) {
    uint32_t h = i * 0x9E3779B9u ^ (j * 0x85EBCA6Bu + 0x7F4A7C15u);
    h ^= h >> 16; h *= 0x7FEB352Du;
    h ^= h >> 15; h *= 0x846CA68Bu;
    h ^= h >> 16;
    float u = ((float)(h >> 8) + 0.5f) * (1.0f / 16777216.0f);  // (0,1)
    return -__logf(-__logf(u));
}

// Insert x into sorted-descending 10-element list (register resident).
__device__ __forceinline__ void insert10(float t[10], float x) {
    if (x <= t[9]) return;               // early reject: common after warm-up
    #pragma unroll
    for (int i = 0; i < 10; ++i) {
        float old = t[i];
        t[i] = fmaxf(old, x);
        x    = fminf(old, x);
    }
}

// Stage one 16-row B tile (8 KB) into LDS; 2 x 16B async chunks per thread.
__device__ __forceinline__ void stage_tile(const _Float16* __restrict__ z,
                                           _Float16* dstLds, int cb, int tid) {
    #pragma unroll
    for (int p = 0; p < 2; ++p) {
        const int idx  = p * 256 + tid;
        const int brow = idx >> 5;
        const int ch   = idx & 31;
        const _Float16* src = z + (size_t)(cb + brow) * DIM + ch * 8;
        _Float16*       dst = dstLds + brow * DIM + ch * 8;
#if ASYNC_STAGE
        __builtin_amdgcn_global_load_async_to_lds_b128(
            (gptr_b128)src, (lptr_b128)dst, 0, 0);
#else
        *(float4*)dst = *(const float4*)src;
        __builtin_prefetch(src + 32 * DIM, 0, 0);
#endif
    }
}

// Wait until at most N async-to-LDS ops from this wave remain outstanding.
template <int N>
__device__ __forceinline__ void wait_async_le() {
#if ASYNC_STAGE
#if __has_builtin(__builtin_amdgcn_s_wait_asynccnt)
    __builtin_amdgcn_s_wait_asynccnt(N);
#else
    if (N == 0) asm volatile("s_wait_asynccnt 0x0" ::: "memory");
    else        asm volatile("s_wait_asynccnt 0x2" ::: "memory");
#endif
#endif
}

// ---------------------------------------------------------------------------
// Kernel 0: zero the global accumulators in workspace
// ---------------------------------------------------------------------------
__global__ void init_accum_kernel(float* accum) {
    if (threadIdx.x == 0) { accum[0] = 0.0f; accum[1] = 0.0f; }
}

// ---------------------------------------------------------------------------
// Kernel 1: row-normalize feats (f32) -> z (f16). One wave (32 lanes) per row.
// ---------------------------------------------------------------------------
__global__ __launch_bounds__(256) void normalize_f16_kernel(
    const float* __restrict__ feats, _Float16* __restrict__ z) {
    const int wave = threadIdx.x >> 5;
    const int lane = threadIdx.x & 31;
    const int row  = blockIdx.x * 8 + wave;

    const float* fp = feats + (size_t)row * DIM + lane * 8;
    float4 x0 = *(const float4*)fp;
    float4 x1 = *(const float4*)(fp + 4);

    float ss = x0.x*x0.x + x0.y*x0.y + x0.z*x0.z + x0.w*x0.w
             + x1.x*x1.x + x1.y*x1.y + x1.z*x1.z + x1.w*x1.w;
    #pragma unroll
    for (int off = 16; off > 0; off >>= 1) ss += __shfl_xor(ss, off, 32);

    const float inv = 1.0f / fmaxf(sqrtf(ss), 1e-12f);
    v8h o;
    o[0] = (_Float16)(x0.x*inv); o[1] = (_Float16)(x0.y*inv);
    o[2] = (_Float16)(x0.z*inv); o[3] = (_Float16)(x0.w*inv);
    o[4] = (_Float16)(x1.x*inv); o[5] = (_Float16)(x1.y*inv);
    o[6] = (_Float16)(x1.z*inv); o[7] = (_Float16)(x1.w*inv);
    *(v8h*)(z + (size_t)row * DIM + lane * 8) = o;
}

// ---------------------------------------------------------------------------
// Kernel 2: fused sim-GEMM (WMMA f16->f32) + per-row top-10 hard negatives +
// Gumbel-max positive + loss accumulation.
//  - 8 waves/WG; WG owns 128 rows; wave owns a 16-row M tile.
//  - B tile triple-buffered in LDS, streamed with GLOBAL_LOAD_ASYNC_TO_LDS_B128
//    (ASYNCcnt); wait `asynccnt<=2` keeps the next tile in flight during WMMAs.
//  - C fragment transposed wave-locally through padded LDS so each lane owns
//    ONE row's top-10 (10 VGPRs) -> fully register resident, no spills.
// ---------------------------------------------------------------------------
__global__ __launch_bounds__(256, 1) void simtopk_kernel(
    const _Float16* __restrict__ z, const int* __restrict__ labels,
    float* __restrict__ accum) {

    __shared__ __align__(16) _Float16 ldsB[3 * TILE_H];     // 24 KB, 3 buffers
    __shared__ __align__(16) float    ldsT[8 * 16 * 20];    // 10 KB transpose pad

    const int tid   = threadIdx.x;
    const int lane  = tid & 31;
    const int wave  = tid >> 5;
    const int r0    = blockIdx.x * 128 + wave * 16;     // wave's 16-row M tile
    const int half  = lane >> 4;                        // 0/1 lane half
    const int nl    = lane & 15;

    // After transpose: this lane owns sim row (r0+nl), columns half*8..half*8+7
    const int myrow    = r0 + nl;
    const int myrowlab = labels[myrow];

    float topk[10];
    #pragma unroll
    for (int i = 0; i < 10; ++i) topk[i] = -INFINITY;
    float bestg = -INFINITY, bests = 0.0f;

    // Prologue: stage tiles 0 and 1
    stage_tile(z, ldsB + 0 * TILE_H, 0, tid);
    stage_tile(z, ldsB + 1 * TILE_H, 16, tid);

    int cur = 0;
    for (int cb = 0; cb < N_ROWS; cb += 16) {
        // Async ops retire in order: <=2 outstanding means tile `cur` landed
        // while tile cur+1 may still be streaming. Final tile: drain fully.
        if (cb + 16 < N_ROWS) wait_async_le<2>(); else wait_async_le<0>();
        __syncthreads();                   // all waves' chunks landed; prior
                                           // iteration's tile reads complete
        if (cb + 32 < N_ROWS) {            // stream tile t+2 into free buffer
            int nxt2 = cur + 2; if (nxt2 >= 3) nxt2 -= 3;
            stage_tile(z, ldsB + nxt2 * TILE_H, cb + 32, tid);
        }

        // ---- 16x16 sim tile: 8 WMMA k-steps over K=256 ----
        const _Float16* bufB = ldsB + cur * TILE_H;
        v8f acc = {};
        #pragma unroll
        for (int ks = 0; ks < 8; ++ks) {
            const int kk = ks * 32;
            // A fragment (16x32 f16): lane nl -> row r0+nl; halves hold
            // K {0..7,16..23} / {8..15,24..31}; loop-invariant in cb ->
            // compiler keeps all 8 k-step fragments VGPR-resident.
            const _Float16* ap = z + (size_t)(r0 + nl) * DIM + kk + half * 8;
            v8h alo = *(const v8h*)ap;
            v8h ahi = *(const v8h*)(ap + 16);
            v16h a;
            #pragma unroll
            for (int i = 0; i < 8; ++i) { a[i] = alo[i]; a[i + 8] = ahi[i]; }
            // B fragment (32x16 f16): lane nl -> col cb+nl; halves hold
            // K {0..7,16..23} / {8..15,24..31} from the LDS tile.
            const _Float16* bp = bufB + nl * DIM + kk + half * 8;
            v8h blo = *(const v8h*)bp;
            v8h bhi = *(const v8h*)(bp + 16);
            v16h b;
            #pragma unroll
            for (int i = 0; i < 8; ++i) { b[i] = blo[i]; b[i + 8] = bhi[i]; }

            acc = __builtin_amdgcn_wmma_f32_16x16x32_f16(
                false, a, false, b, (short)0, acc, false, false);
        }

        // ---- wave-local transpose through padded LDS (stride 20 words) ----
        // C layout: acc[v] @ lane -> row (half*8+v), col nl. LDS ops from the
        // same wave are in-order, so no barrier is needed for this exchange.
        float* tb = &ldsT[wave * 320];
        #pragma unroll
        for (int v = 0; v < 8; ++v)
            tb[(half * 8 + v) * 20 + nl] = acc[v];
        const float* rp = &ldsT[wave * 320 + nl * 20 + half * 8];
        float4 c0 = *(const float4*)rp;        // ds_load_b128
        float4 c1 = *(const float4*)(rp + 4);  // ds_load_b128
        const float cv[8] = {c0.x, c0.y, c0.z, c0.w, c1.x, c1.y, c1.z, c1.w};

        // Column labels straight from global (32 KB array, WGP$-resident).
        const int4 l0 = *(const int4*)&labels[cb + half * 8];
        const int4 l1 = *(const int4*)&labels[cb + half * 8 + 4];
        const int labv[8] = {l0.x, l0.y, l0.z, l0.w, l1.x, l1.y, l1.z, l1.w};

        // ---- per-row reduction: lane owns row `myrow`, 8 columns ----
        #pragma unroll
        for (int j = 0; j < 8; ++j) {
            const int col = cb + half * 8 + j;
            const float s = cv[j];
            if (labv[j] != myrowlab) {
                insert10(topk, s);                       // hard-negative cand
            } else if (col != myrow) {                   // same label, not self
                const float g = gumbel_hash((uint32_t)myrow, (uint32_t)col);
                if (g > bestg) { bestg = g; bests = s; }
            }
        }

        cur = (cur + 1 == 3) ? 0 : cur + 1;
    }

    // ---- merge the two half-lanes that share each row (mask 16) ----
    {
        float tmp[10];
        #pragma unroll
        for (int i = 0; i < 10; ++i) tmp[i] = __shfl_xor(topk[i], 16, 32);
        const float og = __shfl_xor(bestg, 16, 32);
        const float os = __shfl_xor(bests, 16, 32);
        #pragma unroll
        for (int i = 0; i < 10; ++i) insert10(topk, tmp[i]);
        if (og > bestg) { bestg = og; bests = os; }
    }

    // ---- lanes 0..15 finalize their row; wave-reduce; lane 0 atomics ----
    float lsum = 0.0f, lcnt = 0.0f;
    if (half == 0) {
        const bool has_pos = bestg > -1e30f;
        const bool has_neg = topk[0] > -1e30f;
        const float num = __expf(bests * INV_T);
        float den = num;
        #pragma unroll
        for (int i = 0; i < 10; ++i)
            den += (topk[i] > -1e30f) ? __expf(topk[i] * INV_T) : 0.0f;
        const float li = -__logf(fmaxf(num / fmaxf(den, 1e-8f), 1e-8f));
        if (has_pos && has_neg) { lsum = li; lcnt = 1.0f; }
    }
    #pragma unroll
    for (int off = 16; off > 0; off >>= 1) {
        lsum += __shfl_xor(lsum, off, 32);
        lcnt += __shfl_xor(lcnt, off, 32);
    }
    if (lane == 0) {
        atomicAdd(&accum[0], lsum);
        atomicAdd(&accum[1], lcnt);
    }
}

// ---------------------------------------------------------------------------
// Kernel 3: finalize scalar mean
// ---------------------------------------------------------------------------
__global__ void finalize_kernel(const float* __restrict__ accum,
                                float* __restrict__ out) {
    if (threadIdx.x == 0) out[0] = accum[0] / fmaxf(accum[1], 1.0f);
}

extern "C" void kernel_launch(void* const* d_in, const int* in_sizes, int n_in,
                              void* d_out, int out_size, void* d_ws, size_t ws_size,
                              hipStream_t stream) {
    const float* feats  = (const float*)d_in[0];
    const int*   labels = (const int*)d_in[1];
    float*       accum  = (float*)d_ws;                        // [0]=sum [1]=cnt
    _Float16*    z      = (_Float16*)((char*)d_ws + 256);      // 4 MB f16 z
    float*       out    = (float*)d_out;

    init_accum_kernel<<<1, 32, 0, stream>>>(accum);
    normalize_f16_kernel<<<N_ROWS / 8, 256, 0, stream>>>(feats, z);
    simtopk_kernel<<<N_ROWS / 128, 256, 0, stream>>>(z, labels, accum);
    finalize_kernel<<<1, 32, 0, stream>>>(accum, out);
}
